// GraphConvolution_3788161155728
// MI455X (gfx1250) — matrix operand order repaired and checked by reading, not compiled, not run on hardware
//
#include <hip/hip_runtime.h>

// Problem constants (match reference)
#define NNODES 100000
#define NEDGES 3200000
#define NIN    512
#define NOUT   256

typedef __attribute__((ext_vector_type(16))) __bf16 v16bf;
typedef __attribute__((ext_vector_type(8)))  __bf16 v8bf;
typedef __attribute__((ext_vector_type(8)))  float  v8f;
typedef __attribute__((ext_vector_type(4)))  float  v4f;

union BF16x16 { v16bf v; v8bf h[2]; };

// LDS x-tile: 64 rows x 32 bf16 per plane, row stride padded to 80 B
// (20 dwords). 20*r mod 64 is distinct for r=0..15, so the 16-lane
// fragment reads spread across all banks with zero excess conflicts.
#define LDS_ROW_B 80
#define LDS_PLANE (64 * LDS_ROW_B)   // 5120 B per plane (hi, lo)

// ---------------------------------------------------------------------------
// Kernel 0: split W (fp32 [NOUT][NIN]) into bf16 hi/lo planes (512 KB total,
// stays hot in L2 for the GEMM).
// ---------------------------------------------------------------------------
__global__ void __launch_bounds__(256) convert_w(const float* __restrict__ W,
                                                 __bf16* __restrict__ Whi,
                                                 __bf16* __restrict__ Wlo) {
  int i = blockIdx.x * 256 + threadIdx.x;   // NOUT*NIN = 131072 threads exactly
  float w  = W[i];
  __bf16 hi = (__bf16)w;
  Whi[i] = hi;
  Wlo[i] = (__bf16)(w - (float)hi);
}

// ---------------------------------------------------------------------------
// Kernel 1: h = x @ W^T + b via bf16x3 WMMA (hi*hi + hi*lo + lo*hi,
// error ~2^-16 -> near-fp32).
// Block = 256 threads = 8 waves; block tile 64(M) x 256(N) -> x read once.
// Waves: 2 row-groups x 4 col-groups; wave tile 32x64 -> acc[2][4] v8f.
// Per K-chunk of 32: block cooperatively converts the 64x32 fp32 x-slab to
// bf16 hi/lo in LDS (pipelined: next slab's fp32 loads issue right after the
// barrier and land during the 24 WMMAs).
// A frag (16x32): lane-half q holds K = 8q+0..7 (elems 0-7), 16+8q+0..7
//                 (elems 8-15)                         [ISA 7.12.2]
// B frag (32x16): lane-half q holds K = 8q+0..7 (elems 0-7), 16+8q+0..7
//                 (elems 8-15), two 16 B loads from the bf16 W row
// C frag: vgpr r, lane-half q -> M = 8q + r, N = lane&15.
// ---------------------------------------------------------------------------
__global__ void __launch_bounds__(256) gemm_bias(const float* __restrict__ x,
                                                 const __bf16* __restrict__ Whi,
                                                 const __bf16* __restrict__ Wlo,
                                                 const float* __restrict__ b,
                                                 float* __restrict__ hout) {
  __shared__ __align__(16) unsigned char smem[2 * LDS_PLANE];

  const int lane  = threadIdx.x & 31;
  const int wave  = threadIdx.x >> 5;
  const int lhalf = lane >> 4;      // 0/1
  const int lmod  = lane & 15;

  const int rowgroup = wave >> 2;        // 0..1 -> rows +0 / +32
  const int nbase    = (wave & 3) * 64;  // 0,64,128,192
  const int mblock   = blockIdx.x * 64;

  // --- staging role: thread -> (row, 8-float chunk) of the 64x32 slab ---
  const int srow  = threadIdx.x >> 2;        // 0..63
  const int skoff = (threadIdx.x & 3) * 8;   // 0,8,16,24 (floats)
  const int grow  = min(mblock + srow, NNODES - 1);   // clamp partial tile
  const float* xrow = x + (size_t)grow * NIN + skoff;
  const unsigned sstore = (unsigned)(srow * LDS_ROW_B + skoff * 2);

  // --- A-fragment LDS read offsets (chunk1 = +32 B, lo plane = +LDS_PLANE) ---
  unsigned aoff[2];
#pragma unroll
  for (int ms = 0; ms < 2; ++ms)
    aoff[ms] = (unsigned)((rowgroup * 32 + ms * 16 + lmod) * LDS_ROW_B + lhalf * 16);

  v8f acc[2][4] = {};

  // prologue: fetch K-chunk 0 into registers
  v4f r0 = *(const v4f*)(xrow + 0);
  v4f r1 = *(const v4f*)(xrow + 4);

  for (int kc = 0; kc < 16; ++kc) {
    // ---- convert held fp32 regs -> bf16 hi/lo, store to LDS ----
    v8bf hi8, lo8;
#pragma unroll
    for (int j = 0; j < 4; ++j) {
      __bf16 h0 = (__bf16)r0[j];
      hi8[j]     = h0;
      lo8[j]     = (__bf16)(r0[j] - (float)h0);
      __bf16 h1 = (__bf16)r1[j];
      hi8[4 + j] = h1;
      lo8[4 + j] = (__bf16)(r1[j] - (float)h1);
    }
    *(v8bf*)(smem + sstore)             = hi8;
    *(v8bf*)(smem + LDS_PLANE + sstore) = lo8;
    __syncthreads();

    // ---- issue next slab's global loads; they land during the WMMAs ----
    if (kc < 15) {
      const float* xp = xrow + (kc + 1) * 32;
      r0 = *(const v4f*)(xp + 0);
      r1 = *(const v4f*)(xp + 4);
    }

    // ---- A fragments from LDS (conflict-balanced ds_load_b128) ----
    BF16x16 ahi[2], alo[2];
#pragma unroll
    for (int ms = 0; ms < 2; ++ms) {
      ahi[ms].h[0] = *(const v8bf*)(smem + aoff[ms]);
      ahi[ms].h[1] = *(const v8bf*)(smem + aoff[ms] + 32);
      alo[ms].h[0] = *(const v8bf*)(smem + LDS_PLANE + aoff[ms]);
      alo[ms].h[1] = *(const v8bf*)(smem + LDS_PLANE + aoff[ms] + 32);
    }

    // ---- B fragments from L2-resident bf16 W, then bf16x3 WMMAs ----
    const int kbB = kc * 32 + lhalf * 8;
#pragma unroll
    for (int ns = 0; ns < 4; ++ns) {
      const int off = (nbase + ns * 16 + lmod) * NIN + kbB;
      BF16x16 bhi, blo;
      bhi.h[0] = *(const v8bf*)(Whi + off);
      bhi.h[1] = *(const v8bf*)(Whi + off + 16);
      blo.h[0] = *(const v8bf*)(Wlo + off);
      blo.h[1] = *(const v8bf*)(Wlo + off + 16);
#pragma unroll
      for (int ms = 0; ms < 2; ++ms) {
        acc[ms][ns] = __builtin_amdgcn_wmma_f32_16x16x32_bf16(
            false, ahi[ms].v, false, bhi.v, (short)0, acc[ms][ns], false, false);
        acc[ms][ns] = __builtin_amdgcn_wmma_f32_16x16x32_bf16(
            false, ahi[ms].v, false, blo.v, (short)0, acc[ms][ns], false, false);
        acc[ms][ns] = __builtin_amdgcn_wmma_f32_16x16x32_bf16(
            false, alo[ms].v, false, bhi.v, (short)0, acc[ms][ns], false, false);
      }
    }
    __syncthreads();   // LDS slab fully consumed before next overwrite
  }

  // ---- epilogue: + bias, store h ----
#pragma unroll
  for (int ns = 0; ns < 4; ++ns) {
    const int col  = nbase + ns * 16 + lmod;
    const float bv = b[col];
#pragma unroll
    for (int ms = 0; ms < 2; ++ms) {
      const int rowbase = mblock + rowgroup * 32 + ms * 16 + lhalf * 8;
#pragma unroll
      for (int r = 0; r < 8; ++r) {
        const int row = rowbase + r;
        if (row < NNODES)
          hout[(size_t)row * NOUT + col] = acc[ms][ns][r] + bv;
      }
    }
  }
}

// ---------------------------------------------------------------------------
// Kernel 2: zero the output (harness poisons d_out with 0xAA).
// ---------------------------------------------------------------------------
__global__ void __launch_bounds__(256) zero_out(float* __restrict__ out, int n4) {
  int i = blockIdx.x * 256 + threadIdx.x;
  if (i < n4) {
    v4f z = {0.f, 0.f, 0.f, 0.f};
    ((v4f*)out)[i] = z;
  }
}

// ---------------------------------------------------------------------------
// Kernel 3: SpMM scatter. One wave per edge: gather h[col] (1 KB row,
// L2-resident: h = 102 MB < 192 MB L2), scale by val, fp32 hardware atomics
// into out[row] (served by the L2 atomic units).
// ---------------------------------------------------------------------------
__global__ void __launch_bounds__(256) spmm_edges(const int* __restrict__ rows,
                                                  const int* __restrict__ cols,
                                                  const float* __restrict__ vals,
                                                  const float* __restrict__ h,
                                                  float* __restrict__ out) {
  const int lane = threadIdx.x & 31;
  const int e = blockIdx.x * 8 + (threadIdx.x >> 5);
  if (e >= NEDGES) return;

  const int   r = rows[e];
  const int   c = cols[e];
  const float v = vals[e];

  const v4f* hp = (const v4f*)(h + (size_t)c * NOUT);
  float*     op = out + (size_t)r * NOUT;

#pragma unroll
  for (int i = 0; i < 2; ++i) {
    const int q = lane + 32 * i;        // float4 index 0..63 -> 256 cols
    v4f m = hp[q];
#pragma unroll
    for (int j = 0; j < 4; ++j)
      unsafeAtomicAdd(op + 4 * q + j, v * m[j]);   // global_atomic_add_f32
  }
}

// ---------------------------------------------------------------------------
extern "C" void kernel_launch(void* const* d_in, const int* in_sizes, int n_in,
                              void* d_out, int out_size, void* d_ws, size_t ws_size,
                              hipStream_t stream) {
  const float* x        = (const float*)d_in[0];
  const int*   adj_rows = (const int*)  d_in[1];
  const int*   adj_cols = (const int*)  d_in[2];
  const float* adj_vals = (const float*)d_in[3];
  const float* W        = (const float*)d_in[4];
  const float* b        = (const float*)d_in[5];
  float* out = (float*)d_out;

  // Workspace: h [N x NOUT] fp32, then W_hi / W_lo bf16 planes.
  float*  h   = (float*)d_ws;
  __bf16* Whi = (__bf16*)((char*)d_ws + (size_t)NNODES * NOUT * sizeof(float));
  __bf16* Wlo = Whi + (size_t)NOUT * NIN;

  convert_w<<<(NOUT * NIN) / 256, 256, 0, stream>>>(W, Whi, Wlo);

  gemm_bias<<<(NNODES + 63) / 64, 256, 0, stream>>>(x, Whi, Wlo, b, h);

  const int n4 = NNODES * NOUT / 4;
  zero_out<<<(n4 + 255) / 256, 256, 0, stream>>>(out, n4);

  spmm_edges<<<(NEDGES + 7) / 8, 256, 0, stream>>>(adj_rows, adj_cols, adj_vals, h, out);
}
